// Attention_15590731284568
// MI455X (gfx1250) — compile-verified, matrix operand order checked by reading, not run on hardware
//
#include <hip/hip_runtime.h>
#include <hip/hip_bf16.h>
#include <cstdint>
#include <cstddef>

// ---------------------------------------------------------------------------
// Problem constants (reference: B=16, H=W=56, C=768, heads=16, hd=48, ws=7)
// ---------------------------------------------------------------------------
#define NB     16            // batch
#define SEQ    3137          // 1 + 56*56
#define CH     768
#define NHEAD  16
#define HD     48
#define WS7    7
#define WA     49            // window area
#define HWDIM  56
#define NWIN   (NB * 8 * 8)  // 1024 windows
#define QKV_N  2304

typedef __bf16 bf16;
typedef __bf16 v16bf __attribute__((ext_vector_type(16)));
typedef __bf16 v8bf  __attribute__((ext_vector_type(8)));
typedef float  v8f   __attribute__((ext_vector_type(8)));

// ---------------------------------------------------------------------------
// CDNA5 async global->LDS copy (ASYNCcnt-tracked), CDNA5 ISA
// GV mode: per-lane 64-bit global address, per-lane 32-bit LDS byte offset.
// ---------------------------------------------------------------------------
__device__ __forceinline__ void async_ld128(void* lds, const void* g) {
  uint32_t l = (uint32_t)(uintptr_t)lds;       // low 32 bits = LDS offset
  uint64_t a = (uint64_t)(uintptr_t)g;
  asm volatile("global_load_async_to_lds_b128 %0, %1, off"
               :: "v"(l), "v"(a) : "memory");
}
__device__ __forceinline__ void wait_async0() {
  asm volatile("s_wait_asynccnt 0x0" ::: "memory");
}

// ---------------------------------------------------------------------------
// WMMA fragment helpers (CDNA5 16x16x32 bf16 layouts, CDNA5 ISA)
// A-style: lane&15 -> row, lane>>4 selects K-half; element e -> K per table.
// B-style: lane&15 -> col of row-major KxN storage, same K packing.
// ---------------------------------------------------------------------------
__device__ __forceinline__ v16bf frag_row(const bf16* p0, int stride, int lane) {
  int row = lane & 15;
  int kh  = (lane >> 4) << 3;           // 0 or 8
  const bf16* p = p0 + (size_t)row * stride + kh;
  v16bf f;
#pragma unroll
  for (int i = 0; i < 8; ++i) { f[i] = p[i]; f[8 + i] = p[16 + i]; }
  return f;
}

__device__ __forceinline__ v16bf frag_col(const bf16* p0, int stride, int lane) {
  int col = lane & 15;
  int kh  = (lane >> 4) << 3;
  const bf16* p = p0 + (size_t)kh * stride + col;
  v16bf f;
#pragma unroll
  for (int i = 0; i < 8; ++i) {
    f[i]     = p[(size_t)i * stride];
    f[8 + i] = p[(size_t)(16 + i) * stride];
  }
  return f;
}

__device__ __forceinline__ v8f wmma_bf16(v16bf a, v16bf b, v8f c) {
  return __builtin_amdgcn_wmma_f32_16x16x32_bf16(
      /*neg_a=*/false, a, /*neg_b=*/false, b,
      /*c_mod=*/(short)0, c, /*reuse_a=*/false, /*reuse_b=*/false);
}

// ---------------------------------------------------------------------------
// f32 -> bf16 conversion
// ---------------------------------------------------------------------------
__global__ void f32_to_bf16_kernel(const float* __restrict__ src,
                                   bf16* __restrict__ dst, size_t n) {
  size_t i = (size_t)blockIdx.x * blockDim.x + threadIdx.x;
  if (i < n) dst[i] = (bf16)src[i];
}

// ---------------------------------------------------------------------------
// Tiled bf16 WMMA GEMM: C[M,N] = A[M,K] * B[K,N]
// Block 256 threads = 8 wave32 waves; tile 128x128, K-step 32.
// Double-buffered LDS staged with global_load_async_to_lds_b128 so the next
// K-slice streams from HBM while the current one feeds the WMMA pipes.
// Wave (wm in 0..3, wn in 0..1) owns a 32x64 sub-tile = 2x4 WMMA tiles.
// ---------------------------------------------------------------------------
__device__ __forceinline__ void stage_tiles_async(
    const bf16* __restrict__ A, const bf16* __restrict__ Bw,
    bf16 (*As)[32], bf16 (*Bs)[128],
    int m0, int n0, int k0, int M, int N, int K, int tid) {
  // A tile: 128x32 bf16 = 512 x 16B chunks
  for (int i = tid; i < 512; i += 256) {
    int r = i >> 2, kc = (i & 3) << 3;
    int gr = m0 + r;
    if (gr < M)
      async_ld128(&As[r][kc], &A[(size_t)gr * K + k0 + kc]);
  }
  // B tile: 32x128 bf16 = 512 x 16B chunks
  for (int i = tid; i < 512; i += 256) {
    int r = i >> 4, c = (i & 15) << 3;
    async_ld128(&Bs[r][c], &Bw[(size_t)(k0 + r) * N + n0 + c]);
  }
}

template <bool OUT_F32>
__global__ __launch_bounds__(256)
void gemm_bf16_kernel(const bf16* __restrict__ A, const bf16* __restrict__ Bw,
                      const float* __restrict__ bias,
                      float* __restrict__ Cf, bf16* __restrict__ Cb,
                      int M, int N, int K) {
  __shared__ bf16 As[2][128][32];
  __shared__ bf16 Bs[2][32][128];

  const int tid  = threadIdx.x;
  const int lane = tid & 31;
  const int w    = tid >> 5;
  const int wm   = w >> 1;   // 0..3
  const int wn   = w & 1;    // 0..1
  const int m0   = blockIdx.y * 128;
  const int n0   = blockIdx.x * 128;

  v8f acc[2][4] = {};

  const int nk = K >> 5;
  stage_tiles_async(A, Bw, As[0], Bs[0], m0, n0, 0, M, N, K, tid);
  wait_async0();
  __syncthreads();

  for (int t = 0; t < nk; ++t) {
    const int cur = t & 1;
    if (t + 1 < nk)  // prefetch next K-slice into the other buffer
      stage_tiles_async(A, Bw, As[cur ^ 1], Bs[cur ^ 1],
                        m0, n0, (t + 1) << 5, M, N, K, tid);

    v16bf af[2];
#pragma unroll
    for (int mt = 0; mt < 2; ++mt)
      af[mt] = frag_row(&As[cur][wm * 32 + mt * 16][0], 32, lane);
#pragma unroll
    for (int nt = 0; nt < 4; ++nt) {
      v16bf bfv = frag_col(&Bs[cur][0][wn * 64 + nt * 16], 128, lane);
#pragma unroll
      for (int mt = 0; mt < 2; ++mt)
        acc[mt][nt] = wmma_bf16(af[mt], bfv, acc[mt][nt]);
    }

    wait_async0();      // next buffer resident
    __syncthreads();    // everyone done reading cur before it is re-staged
  }

  const int cn = lane & 15;
  const int rb = (lane >> 4) << 3;
#pragma unroll
  for (int mt = 0; mt < 2; ++mt) {
#pragma unroll
    for (int nt = 0; nt < 4; ++nt) {
      int gn = n0 + wn * 64 + nt * 16 + cn;
#pragma unroll
      for (int r = 0; r < 8; ++r) {
        int gm = m0 + wm * 32 + mt * 16 + rb + r;
        if (gm < M) {
          float v = acc[mt][nt][r];
          if (OUT_F32) Cf[(size_t)gm * N + gn] = v + (bias ? bias[gn] : 0.0f);
          else         Cb[(size_t)gm * N + gn] = (bf16)v;
        }
      }
    }
  }
}

// ---------------------------------------------------------------------------
// Class-token attention: x_cls[b,h,:] = softmax(q_cls . k_img * s) @ v_img
// One block per (b, head), two-pass softmax, LDS reductions.
// ---------------------------------------------------------------------------
__global__ __launch_bounds__(256)
void cls_attn_kernel(const bf16* __restrict__ qkv, float* __restrict__ x_cls) {
  const int b = blockIdx.x, h = blockIdx.y, tid = threadIdx.x;
  __shared__ float qsh[HD];
  __shared__ float red[256];
  __shared__ float accs[HD];
  __shared__ float lsum;

  const size_t qoff = ((size_t)b * SEQ) * QKV_N + h * HD;  // which=0, n=0
  if (tid < HD) qsh[tid] = (float)qkv[qoff + tid];
  if (tid < HD) accs[tid] = 0.0f;
  if (tid == 0) lsum = 0.0f;
  __syncthreads();

  const float scale = 0.14433756729740643f;  // 48^-0.5

  float mloc = -1e30f;
  for (int t = tid; t < HWDIM * HWDIM; t += 256) {
    size_t koff = ((size_t)b * SEQ + 1 + t) * QKV_N + CH + h * HD;
    float s = 0.0f;
    for (int d = 0; d < HD; ++d) s += qsh[d] * (float)qkv[koff + d];
    mloc = fmaxf(mloc, s * scale);
  }
  red[tid] = mloc;
  __syncthreads();
  for (int st = 128; st > 0; st >>= 1) {
    if (tid < st) red[tid] = fmaxf(red[tid], red[tid + st]);
    __syncthreads();
  }
  const float m = red[0];
  __syncthreads();

  float lloc = 0.0f;
  float aloc[HD];
  for (int d = 0; d < HD; ++d) aloc[d] = 0.0f;
  for (int t = tid; t < HWDIM * HWDIM; t += 256) {
    size_t koff = ((size_t)b * SEQ + 1 + t) * QKV_N + CH + h * HD;
    float s = 0.0f;
    for (int d = 0; d < HD; ++d) s += qsh[d] * (float)qkv[koff + d];
    float p = __expf(s * scale - m);
    lloc += p;
    const bf16* vp = qkv + koff + CH;  // which=2
    for (int d = 0; d < HD; ++d) aloc[d] += p * (float)vp[d];
  }
  atomicAdd(&lsum, lloc);
  for (int d = 0; d < HD; ++d) atomicAdd(&accs[d], aloc[d]);
  __syncthreads();
  if (tid < HD) x_cls[(size_t)b * CH + h * HD + tid] = accs[tid] / lsum;
}

// ---------------------------------------------------------------------------
// v_cls = x_cls @ kv_global_w[:, C:2C]   (softmax over 1 key == 1, so the
// global-broadcast term is exactly v_cls; k_cls is never needed)
// ---------------------------------------------------------------------------
__global__ void vcls_kernel(const float* __restrict__ x_cls,
                            const float* __restrict__ kvg_w,
                            float* __restrict__ v_cls) {
  int i = blockIdx.x * blockDim.x + threadIdx.x;
  if (i >= NB * CH) return;
  int b = i / CH, c = i % CH;
  float s = 0.0f;
  for (int k = 0; k < CH; ++k)
    s += x_cls[(size_t)b * CH + k] * kvg_w[(size_t)k * (2 * CH) + CH + c];
  v_cls[i] = s;
}

__global__ void token0_kernel(const float* __restrict__ x_cls,
                              bf16* __restrict__ Y) {
  int i = blockIdx.x * blockDim.x + threadIdx.x;
  if (i >= NB * CH) return;
  int b = i / CH, c = i % CH;
  Y[(size_t)b * SEQ * CH + c] = (bf16)x_cls[i];
}

// ---------------------------------------------------------------------------
// Window attention, one block per (window, head). 4 wave32 waves.
// 49x49 attention padded to 64x64; both GEMMs on WMMA bf16; fp32 softmax.
// Q/K/V staged via async global->LDS copies into zero-filled padded tiles.
// Fused epilogue: + v_cls broadcast, bf16 store into concat buffer Y.
// ---------------------------------------------------------------------------
__global__ __launch_bounds__(128)
void win_attn_kernel(const bf16* __restrict__ qkv,
                     const float* __restrict__ rel_table,
                     const float* __restrict__ v_cls,
                     bf16* __restrict__ Y) {
  const int win = blockIdx.x, h = blockIdx.y;
  const int b = win >> 6, wloc = win & 63;
  const int wy = wloc >> 3, wx = wloc & 7;
  const int tid = threadIdx.x;
  const int lane = tid & 31;
  const int w = tid >> 5;  // wave 0..3 -> query rows [16w, 16w+16)

  __shared__ bf16  Qs[64][64];
  __shared__ bf16  Ks[64][64];
  __shared__ bf16  Vs[64][64];
  __shared__ float Sf[64][64];
  __shared__ bf16  Ps[64][64];

  // zero-fill padded tiles (padding feeds WMMA K-dim, must be 0)
  {
    v8bf z = {};
    for (int i = tid; i < (64 * 64) / 8; i += 128) {
      ((v8bf*)Qs)[i] = z; ((v8bf*)Ks)[i] = z; ((v8bf*)Vs)[i] = z;
    }
  }
  __syncthreads();

  // async-copy the valid 49x48 region: 49 rows * 6 x 16B chunks per tensor
  for (int i = tid; i < WA * (HD / 8); i += 128) {
    int r = i / (HD / 8);
    int dc = (i % (HD / 8)) << 3;
    int y = wy * WS7 + r / WS7;
    int x = wx * WS7 + r % WS7;
    size_t base = ((size_t)b * SEQ + 1 + y * HWDIM + x) * QKV_N + h * HD + dc;
    async_ld128(&Qs[r][dc], qkv + base);
    async_ld128(&Ks[r][dc], qkv + base + CH);
    async_ld128(&Vs[r][dc], qkv + base + 2 * CH);
  }
  wait_async0();
  __syncthreads();

  // S = Q . K^T  (K-dim 48 zero-padded to 64 -> two 32-steps)
  v8f sacc[4] = {};
#pragma unroll
  for (int kk = 0; kk < 64; kk += 32) {
    v16bf a = frag_row(&Qs[w * 16][kk], 64, lane);
#pragma unroll
    for (int nt = 0; nt < 4; ++nt) {
      // B[k][n] = Ks[n][k]  -> A-style read of Ks gives the transposed operand
      v16bf bfv = frag_row(&Ks[nt * 16][kk], 64, lane);
      sacc[nt] = wmma_bf16(a, bfv, sacc[nt]);
    }
  }

  const float scale = 0.14433756729740643f;
  {
    const int cn = lane & 15;
    const int rb = (lane >> 4) << 3;
#pragma unroll
    for (int nt = 0; nt < 4; ++nt) {
      int n = nt * 16 + cn;
#pragma unroll
      for (int r = 0; r < 8; ++r) {
        int qr = w * 16 + rb + r;
        float s = sacc[nt][r] * scale;
        if (qr < WA && n < WA) {
          int dy = qr / WS7 - n / WS7 + (WS7 - 1);
          int dx = qr % WS7 - n % WS7 + (WS7 - 1);
          s += rel_table[(dy * (2 * WS7 - 1) + dx) * NHEAD + h];
        } else {
          s = -1e30f;  // mask padded keys / rows
        }
        Sf[qr][n] = s;
      }
    }
  }
  __syncthreads();

  // fp32 row softmax -> bf16 P
  if (tid < 64) {
    float m = -1e30f;
    for (int n = 0; n < WA; ++n) m = fmaxf(m, Sf[tid][n]);
    float l = 0.0f;
    for (int n = 0; n < WA; ++n) l += __expf(Sf[tid][n] - m);
    float inv = 1.0f / l;
    for (int n = 0; n < WA; ++n) Ps[tid][n] = (bf16)(__expf(Sf[tid][n] - m) * inv);
    for (int n = WA; n < 64; ++n) Ps[tid][n] = (bf16)0.0f;
  }
  __syncthreads();

  // O = P . V  (64x64 @ 64x48)
  v8f oacc[3] = {};
#pragma unroll
  for (int kk = 0; kk < 64; kk += 32) {
    v16bf a = frag_row(&Ps[w * 16][kk], 64, lane);
#pragma unroll
    for (int nt = 0; nt < 3; ++nt) {
      v16bf bfv = frag_col(&Vs[kk][nt * 16], 64, lane);
      oacc[nt] = wmma_bf16(a, bfv, oacc[nt]);
    }
  }

  const int cn = lane & 15;
  const int rb = (lane >> 4) << 3;
#pragma unroll
  for (int nt = 0; nt < 3; ++nt) {
    int d = nt * 16 + cn;
#pragma unroll
    for (int r = 0; r < 8; ++r) {
      int qr = w * 16 + rb + r;
      if (qr < WA && d < HD) {
        int y = wy * WS7 + qr / WS7;
        int x = wx * WS7 + qr % WS7;
        float val = oacc[nt][r] + v_cls[(size_t)b * CH + h * HD + d];
        Y[((size_t)b * SEQ + 1 + y * HWDIM + x) * CH + h * HD + d] = (bf16)val;
      }
    }
  }
}

// ---------------------------------------------------------------------------
// Host-side orchestration
// ---------------------------------------------------------------------------
extern "C" void kernel_launch(void* const* d_in, const int* in_sizes, int n_in,
                              void* d_out, int out_size, void* d_ws, size_t ws_size,
                              hipStream_t stream) {
  const float* x      = (const float*)d_in[0];
  const float* qkv_w  = (const float*)d_in[1];
  const float* kvg_w  = (const float*)d_in[2];
  const float* proj_w = (const float*)d_in[3];
  const float* proj_b = (const float*)d_in[4];
  const float* rel_t  = (const float*)d_in[5];
  float* out = (float*)d_out;

  const int M = NB * SEQ;  // 50192

  char* ws = (char*)d_ws;
  size_t off = 0;
  auto carve = [&](size_t bytes) -> void* {
    void* p = ws + off;
    off = (off + bytes + 255) & ~(size_t)255;
    return p;
  };
  bf16*  Xb    = (bf16*)carve((size_t)M * CH * sizeof(bf16));
  bf16*  Qwb   = (bf16*)carve((size_t)CH * QKV_N * sizeof(bf16));
  bf16*  Pwb   = (bf16*)carve((size_t)CH * CH * sizeof(bf16));
  bf16*  QKVb  = (bf16*)carve((size_t)M * QKV_N * sizeof(bf16));
  bf16*  Ycat  = (bf16*)carve((size_t)M * CH * sizeof(bf16));
  float* x_cls = (float*)carve((size_t)NB * CH * sizeof(float));
  float* v_cls = (float*)carve((size_t)NB * CH * sizeof(float));
  (void)ws_size; (void)in_sizes; (void)n_in; (void)out_size;

  // 1) bf16 casts
  {
    size_t n = (size_t)M * CH;
    f32_to_bf16_kernel<<<dim3((unsigned)((n + 255) / 256)), 256, 0, stream>>>(x, Xb, n);
  }
  {
    size_t n = (size_t)CH * QKV_N;
    f32_to_bf16_kernel<<<dim3((unsigned)((n + 255) / 256)), 256, 0, stream>>>(qkv_w, Qwb, n);
  }
  {
    size_t n = (size_t)CH * CH;
    f32_to_bf16_kernel<<<dim3((unsigned)((n + 255) / 256)), 256, 0, stream>>>(proj_w, Pwb, n);
  }

  // 2) qkv = x @ qkv_w  (bf16 out)
  {
    dim3 grid(QKV_N / 128, (M + 127) / 128);
    gemm_bf16_kernel<false><<<grid, 256, 0, stream>>>(Xb, Qwb, nullptr,
                                                      nullptr, QKVb, M, QKV_N, CH);
  }

  // 3) class-token attention -> x_cls (fp32)
  cls_attn_kernel<<<dim3(NB, NHEAD), 256, 0, stream>>>(QKVb, x_cls);

  // 4) v_cls (global-broadcast term; softmax over the single global key == 1)
  vcls_kernel<<<(NB * CH + 255) / 256, 256, 0, stream>>>(x_cls, kvg_w, v_cls);

  // 5) window attention + fused v_cls add -> Ycat tokens 1..3136
  win_attn_kernel<<<dim3(NWIN, NHEAD), 128, 0, stream>>>(QKVb, rel_t, v_cls, Ycat);

  // 6) token 0 of concat buffer
  token0_kernel<<<(NB * CH + 255) / 256, 256, 0, stream>>>(x_cls, Ycat);

  // 7) out = Ycat @ proj_w + proj_b  (fp32 out)
  {
    dim3 grid(CH / 128, (M + 127) / 128);
    gemm_bf16_kernel<true><<<grid, 256, 0, stream>>>(Ycat, Pwb, proj_b,
                                                     out, nullptr, M, CH, CH);
  }
}
